// MaxSimREModel_16509854285968
// MI455X (gfx1250) — matrix operand order repaired and checked by reading, not compiled, not run on hardware
//
#include <hip/hip_runtime.h>
#include <hip/hip_bf16.h>

typedef __attribute__((ext_vector_type(16))) _Float16 v16h;
typedef __attribute__((ext_vector_type(8)))  float    v8f;

#define HID   768
#define MDIM  128
#define LTOK  512
#define NBAT  64
#define PAD   8
#define QSTR  (MDIM + PAD)   // 136 halves = 272 B row stride (16B aligned, bank stride 68)

// ---------------------------------------------------------------------------
// 16-byte global->LDS copy via CDNA5 async-load-to-LDS (ASYNCcnt-tracked).
// Probe-verified signature: (v4i addrspace(1)*, v4i addrspace(3)*, Ii, Ii)
// ---------------------------------------------------------------------------
#if defined(__has_builtin)
#if __has_builtin(__builtin_amdgcn_global_load_async_to_lds_b128)
#define HAVE_ASYNC_LDS 1
#endif
#endif

#define AS1 __attribute__((address_space(1)))
#define AS3 __attribute__((address_space(3)))

typedef int v4i __attribute__((vector_size(16)));
typedef AS1 v4i* gv4i_ptr;   // global-AS pointer to int4
typedef AS3 v4i* lv4i_ptr;   // LDS-AS pointer to int4

__device__ __forceinline__ void copy16_g2l(void* l, const void* g) {
#ifdef HAVE_ASYNC_LDS
  __builtin_amdgcn_global_load_async_to_lds_b128(
      (gv4i_ptr)(unsigned long long)(uintptr_t)g,
      (lv4i_ptr)(unsigned long long)(uintptr_t)l, 0, 0);
#else
  *(uint4*)l = *(const uint4*)g;
#endif
}

__device__ __forceinline__ void async_copy_wait() {
#ifdef HAVE_ASYNC_LDS
  asm volatile("s_wait_asynccnt 0x0" ::: "memory");
#endif
}

// ---- WMMA fragment loaders (CDNA5 16-bit layouts, wave32) -------------------
// A matrix 16x32 f16: lane holds row M=lane&15; VGPR v holds K pair at
//   K = 2*(v&3) + 16*(v>>2) + 8*(lane>=16)
__device__ __forceinline__ v16h load_a_frag(const _Float16* rowp, int h) {
  v16h f;
#pragma unroll
  for (int v = 0; v < 8; ++v) {
    int k = ((v & 3) << 1) + ((v >> 2) << 4) + (h << 3);
    f[2 * v]     = rowp[k];
    f[2 * v + 1] = rowp[k + 1];
  }
  return f;
}
// B matrix 32x16 f16: lane holds col N=lane&15; VGPR v holds K pair at
//   K = 2*(v&3) + 16*(v>>2) + 8*(lane>=16)
__device__ __forceinline__ v16h load_b_frag(const _Float16* colp, int h) {
  v16h f;
#pragma unroll
  for (int v = 0; v < 8; ++v) {
    int k = ((v & 3) << 1) + ((v >> 2) << 4) + (h << 3);
    f[2 * v]     = colp[k];
    f[2 * v + 1] = colp[k + 1];
  }
  return f;
}

// ---- Pass 0: Wt_g[n][k] = (f16) W[k][n]  (128 x 768 f16, 192 KB, L2-resident)
__global__ __launch_bounds__(256) void transpose_w_f16(
    const float* __restrict__ W, _Float16* __restrict__ Wg) {
  int gid = blockIdx.x * 256 + threadIdx.x;
  for (int e = gid; e < MDIM * HID; e += (int)gridDim.x * 256) {
    int n = e / HID;
    int k = e - n * HID;
    Wg[e] = (_Float16)W[(size_t)k * MDIM + n];
  }
}

// ---- Pass A: D = l2norm(H @ W), stored f16 [B*L, 128] -----------------------
// Grid: 512 WGs x 256 threads. Each WG: 64 tokens x 128 cols, K=768 in 6x128.
__global__ __launch_bounds__(256) void proj_l2norm_f16(
    const float* __restrict__ H, const _Float16* __restrict__ Wg,
    _Float16* __restrict__ D) {
  __shared__ _Float16 Hs[64][QSTR];    // token-major f16 tile of H chunk
  __shared__ _Float16 Wt[128][QSTR];   // W chunk, n-major: Wt[n][k]
  __shared__ float    red[2][64];      // cross-wave ssq partials

  const int tid  = threadIdx.x;
  const int wave = tid >> 5;
  const int lane = tid & 31;
  const int l15  = lane & 15;
  const int h    = lane >> 4;
  const int rb   = wave & 3;           // row (token) block 0..3
  const int ch   = wave >> 2;          // col half 0..1 (64 cols each)
  const size_t tileBase = (size_t)blockIdx.x * 64;

  v8f acc[4] = {};

  for (int kc = 0; kc < 6; ++kc) {
    // 1) kick off async DMA of the W chunk first (hides under conversion work)
    {
      int n  = tid >> 1;
      int c0 = (tid & 1) * 64;
      const _Float16* src = Wg + (size_t)n * HID + kc * 128 + c0;
#pragma unroll
      for (int i = 0; i < 8; ++i)
        copy16_g2l(&Wt[n][c0 + 8 * i], src + 8 * i);
    }
    // 2) stage H tile chunk: 64 x 128 f32 -> f16, packed 8-half (b128) stores
    {
      int row = tid & 63;
      int c0  = (tid >> 6) * 32;
      const float* src = H + (tileBase + row) * HID + kc * 128 + c0;
      if (kc + 1 < 6) __builtin_prefetch(src + 128, 0, 3);
#pragma unroll
      for (int i = 0; i < 4; ++i) {
        float4 f0 = ((const float4*)src)[2 * i];
        float4 f1 = ((const float4*)src)[2 * i + 1];
        union { _Float16 hh[8]; uint4 u; } p;
        p.hh[0] = (_Float16)f0.x; p.hh[1] = (_Float16)f0.y;
        p.hh[2] = (_Float16)f0.z; p.hh[3] = (_Float16)f0.w;
        p.hh[4] = (_Float16)f1.x; p.hh[5] = (_Float16)f1.y;
        p.hh[6] = (_Float16)f1.z; p.hh[7] = (_Float16)f1.w;
        *(uint4*)&Hs[row][c0 + 8 * i] = p.u;
      }
    }
    async_copy_wait();
    __syncthreads();
#pragma unroll
    for (int ks = 0; ks < 4; ++ks) {
      // preload A + all 4 B fragments, then issue 4 back-to-back WMMAs
      v16h a = load_a_frag(&Hs[rb * 16 + l15][ks * 32], h);
      v16h bf[4];
#pragma unroll
      for (int cb = 0; cb < 4; ++cb)
        bf[cb] = load_b_frag(&Wt[ch * 64 + cb * 16 + l15][ks * 32], h);
#pragma unroll
      for (int cb = 0; cb < 4; ++cb)
        acc[cb] = __builtin_amdgcn_wmma_f32_16x16x32_f16(
            false, a, false, bf[cb], (short)0, acc[cb], false, false);
    }
    __syncthreads();
  }

  // per-token sum of squares: this wave covers 64 of the 128 m-columns
  float p[8];
#pragma unroll
  for (int r = 0; r < 8; ++r) {
    float s = 0.f;
#pragma unroll
    for (int cb = 0; cb < 4; ++cb) { float v = acc[cb][r]; s += v * v; }
#pragma unroll
    for (int m = 1; m <= 8; m <<= 1) s += __shfl_xor(s, m, 32);
    p[r] = s;
  }
  if (l15 == 0) {
#pragma unroll
    for (int r = 0; r < 8; ++r) red[ch][rb * 16 + r + 8 * h] = p[r];
  }
  __syncthreads();

#pragma unroll
  for (int r = 0; r < 8; ++r) {
    int rowLocal = rb * 16 + r + 8 * h;
    float ssq = red[0][rowLocal] + red[1][rowLocal];
    float inv = 1.0f / fmaxf(sqrtf(ssq), 1e-12f);
    size_t tok = tileBase + (size_t)rowLocal;
#pragma unroll
    for (int cb = 0; cb < 4; ++cb) {
      int m = ch * 64 + cb * 16 + l15;
      D[tok * MDIM + m] = (_Float16)(acc[cb][r] * inv);
    }
  }
}

// ---- Pass B: Q build + MaxSim + gate + sum over P ---------------------------
// Grid: (4, B) x 256 threads. Each WG: 128 probe rows, all 512 tokens.
// D[b] is streamed in 16 chunks of 32 tokens, double-buffered via async DMA.
__global__ __launch_bounds__(256) void maxsim_logits(
    const _Float16* __restrict__ D,
    const int* __restrict__ ctx, const int* __restrict__ btw,
    const int* __restrict__ st, const int* __restrict__ ot,
    const int* __restrict__ sf,
    const float* __restrict__ lp, const float* __restrict__ sW,
    const float* __restrict__ oW, const float* __restrict__ dW,
    float* __restrict__ out) {
  __shared__ _Float16 Qs[128][QSTR];
  __shared__ _Float16 Ds[2][32][QSTR];
  __shared__ unsigned char evid[LTOK];
  __shared__ int flag;

  const int tid  = threadIdx.x;
  const int wave = tid >> 5;
  const int lane = tid & 31;
  const int l15  = lane & 15;
  const int h    = lane >> 4;
  const int b    = blockIdx.y;
  const int quad = blockIdx.x;   // which 128-row quarter of the 512 probes

  if (tid == 0) flag = 0;
  __syncthreads();
  int loc = btw[b * LTOK + tid] | btw[b * LTOK + tid + 256];
  if (loc) atomicOr(&flag, 1);
  __syncthreads();
  const int* msk = flag ? btw : ctx;
  evid[tid]       = (unsigned char)(msk[b * LTOK + tid] != 0);
  evid[tid + 256] = (unsigned char)(msk[b * LTOK + tid + 256] != 0);

  // stage chunk c (32 tokens x 128 halves) into buffer buf
  const int srow = tid >> 3;          // 32 rows, 8 threads per row
  const int scol = (tid & 7) * 16;    // 16 halves per thread
#define STAGE_D(c, buf)                                                       \
  {                                                                           \
    const _Float16* src =                                                     \
        D + ((size_t)b * LTOK + (c) * 32 + srow) * MDIM + scol;               \
    copy16_g2l(&Ds[buf][srow][scol], src);                                    \
    copy16_g2l(&Ds[buf][srow][scol + 8], src + 8);                            \
  }

  // prime the pipeline with chunk 0 while building Q
  STAGE_D(0, 0);

  // build + l2norm the 128 conditioned probes for this WG (threads 0..127)
  if (tid < 128) {
    int row = quad * 128 + tid;
    const float* lpr = lp + (size_t)row * MDIM;
    const float* sv  = sW + (size_t)st[b] * MDIM;
    const float* ov  = oW + (size_t)ot[b] * MDIM;
    const float* dv  = dW + (size_t)sf[b] * MDIM;
    float ssq = 0.f;
    for (int m = 0; m < MDIM; ++m) {
      float t = lpr[m] + sv[m] + ov[m] + dv[m];
      ssq += t * t;
    }
    float inv = 1.0f / fmaxf(sqrtf(ssq), 1e-12f);
    for (int m = 0; m < MDIM; m += 2) {
      float t0 = (lpr[m] + sv[m] + ov[m] + dv[m]) * inv;
      float t1 = (lpr[m + 1] + sv[m + 1] + ov[m + 1] + dv[m + 1]) * inv;
      union { _Float16 hh[2]; unsigned u; } p;
      p.hh[0] = (_Float16)t0; p.hh[1] = (_Float16)t1;
      *(unsigned*)&Qs[tid][m] = p.u;
    }
  }
  async_copy_wait();
  __syncthreads();

  // A fragments (this wave's 16 probe rows), kept resident across all tokens
  v16h qf[4];
#pragma unroll
  for (int ks = 0; ks < 4; ++ks)
    qf[ks] = load_a_frag(&Qs[wave * 16 + l15][ks * 32], h);

  float rmax[8];
#pragma unroll
  for (int r = 0; r < 8; ++r) rmax[r] = -3.0e38f;

  for (int tc = 0; tc < 16; ++tc) {
    const int cur = tc & 1;
    // issue DMA for next chunk into the other buffer, then compute on current
    if (tc + 1 < 16) STAGE_D(tc + 1, cur ^ 1);
#pragma unroll
    for (int tt = 0; tt < 2; ++tt) {
      // preload all 4 B fragments, then 4 back-to-back WMMAs
      v16h bf[4];
#pragma unroll
      for (int ks = 0; ks < 4; ++ks)
        bf[ks] = load_b_frag(&Ds[cur][tt * 16 + l15][ks * 32], h);
      v8f acc = {};
#pragma unroll
      for (int ks = 0; ks < 4; ++ks)
        acc = __builtin_amdgcn_wmma_f32_16x16x32_f16(
            false, qf[ks], false, bf[ks], (short)0, acc, false, false);
      int tok = tc * 32 + tt * 16 + l15;   // this lane's token column
      bool ev = evid[tok] != 0;
#pragma unroll
      for (int r = 0; r < 8; ++r) {
        float v = ev ? acc[r] : -10000.0f;
        rmax[r] = fmaxf(rmax[r], v);
      }
    }
    async_copy_wait();
    __syncthreads();
  }
#undef STAGE_D

  // max across the 16 token-columns held by this half-wave
#pragma unroll
  for (int r = 0; r < 8; ++r) {
#pragma unroll
    for (int m = 1; m <= 8; m <<= 1)
      rmax[r] = fmaxf(rmax[r], __shfl_xor(rmax[r], m, 32));
  }
  // lanes 0 and 16 each own 8 consecutive probe rows = one (r, p=0..7) group
  if (l15 == 0) {
    float s = 0.f;
#pragma unroll
    for (int r = 0; r < 8; ++r) {
      float v = rmax[r];
      s += (v > -1000.0f) ? v : 0.0f;
    }
    int rlab = ((quad * 128 + wave * 16) >> 3) + h;
    out[b * 64 + rlab] = s;
  }
}

extern "C" void kernel_launch(void* const* d_in, const int* in_sizes, int n_in,
                              void* d_out, int out_size, void* d_ws, size_t ws_size,
                              hipStream_t stream) {
  const float* H   = (const float*)d_in[0];
  const int*   ctx = (const int*)d_in[1];
  const int*   btw = (const int*)d_in[2];
  const int*   st  = (const int*)d_in[3];
  const int*   ot  = (const int*)d_in[4];
  const int*   sf  = (const int*)d_in[5];
  const float* W   = (const float*)d_in[6];
  const float* lp  = (const float*)d_in[7];
  const float* sW  = (const float*)d_in[8];
  const float* oW  = (const float*)d_in[9];
  const float* dW  = (const float*)d_in[10];
  float* out = (float*)d_out;

  // workspace layout: D intermediate f16 (8 MB) | transposed W f16 (192 KB)
  _Float16* Dbuf = (_Float16*)d_ws;
  _Float16* Wg   = (_Float16*)((char*)d_ws +
                               (size_t)NBAT * LTOK * MDIM * sizeof(_Float16));

  hipLaunchKernelGGL(transpose_w_f16, dim3(96), dim3(256), 0, stream, W, Wg);
  hipLaunchKernelGGL(proj_l2norm_f16, dim3((NBAT * LTOK) / 64), dim3(256), 0,
                     stream, H, Wg, Dbuf);
  hipLaunchKernelGGL(maxsim_logits, dim3(4, NBAT), dim3(256), 0, stream,
                     Dbuf, ctx, btw, st, ot, sf, lp, sW, oW, dW, out);
}
